// GNN_18803366821915
// MI455X (gfx1250) — matrix operand order repaired and checked by reading, not compiled, not run on hardware
//
#include <hip/hip_runtime.h>
#include <hip/hip_bf16.h>

#define N_NODES 100000
#define E_EDGES 1600000
#define C_DIM   128
#define NEG_SLOPE 0.2f

typedef __attribute__((ext_vector_type(16))) _Float16 v16h;
typedef __attribute__((ext_vector_type(8)))  float    v8f;

// ---------------- init ----------------
__global__ void init_maxkey_k(unsigned int* __restrict__ mk, int n) {
    int i = blockIdx.x * blockDim.x + threadIdx.x;
    if (i < n) mk[i] = 0x007FFFFFu;   // monotonic key of -inf
}

// ---------------- node projection: xl = x@Wl+bl, xr = x@Wr+br ----------------
__global__ void node_proj_k(const float* __restrict__ x,
                            const float* __restrict__ Wl, const float* __restrict__ bl,
                            const float* __restrict__ Wr, const float* __restrict__ br,
                            float* __restrict__ xl, float* __restrict__ xr) {
    int n = blockIdx.x;
    int c = threadIdx.x;              // 0..127
    __shared__ float xs[8];
    if (c < 8) xs[c] = x[n * 8 + c];
    __syncthreads();
    float al = bl[c], ar = br[c];
#pragma unroll
    for (int k = 0; k < 8; ++k) {
        float xv = xs[k];
        al = fmaf(xv, Wl[k * C_DIM + c], al);
        ar = fmaf(xv, Wr[k * C_DIM + c], ar);
    }
    xl[n * C_DIM + c] = al;
    xr[n * C_DIM + c] = ar;
}

__device__ __forceinline__ float wave_sum(float v) {
#pragma unroll
    for (int off = 16; off > 0; off >>= 1) v += __shfl_xor(v, off, 32);
    return v;
}

// ---------------- pass A: per-edge logits + segment max ----------------
__global__ void edge_logits_k(const int* __restrict__ ei, const float* __restrict__ ef,
                              const float* __restrict__ xl, const float* __restrict__ xr,
                              const float* __restrict__ We, const float* __restrict__ att,
                              float* __restrict__ logits, unsigned int* __restrict__ maxkey) {
    int w = blockIdx.x * (blockDim.x >> 5) + (threadIdx.x >> 5);
    if (w >= E_EDGES) return;
    int lane = threadIdx.x & 31;
    int src = ei[w];
    int dst = ei[E_EDGES + w];
    float e0 = ef[2 * w], e1 = ef[2 * w + 1];
    float4 a  = ((const float4*)(xl + (size_t)src * C_DIM))[lane];
    float4 b  = ((const float4*)(xr + (size_t)dst * C_DIM))[lane];
    float4 w0 = ((const float4*)We)[lane];
    float4 w1 = ((const float4*)(We + C_DIM))[lane];
    float4 at = ((const float4*)att)[lane];
    float acc = 0.f, m;
    m = a.x + b.x + e0 * w0.x + e1 * w1.x; m = m > 0.f ? m : NEG_SLOPE * m; acc = fmaf(m, at.x, acc);
    m = a.y + b.y + e0 * w0.y + e1 * w1.y; m = m > 0.f ? m : NEG_SLOPE * m; acc = fmaf(m, at.y, acc);
    m = a.z + b.z + e0 * w0.z + e1 * w1.z; m = m > 0.f ? m : NEG_SLOPE * m; acc = fmaf(m, at.z, acc);
    m = a.w + b.w + e0 * w0.w + e1 * w1.w; m = m > 0.f ? m : NEG_SLOPE * m; acc = fmaf(m, at.w, acc);
    acc = wave_sum(acc);
    if (lane == 0) {
        logits[w] = acc;
        unsigned ib = __float_as_uint(acc);
        unsigned key = (ib & 0x80000000u) ? ~ib : (ib | 0x80000000u);
        atomicMax(&maxkey[dst], key);
    }
}

// ---------------- pass B: exp + segment sum ----------------
__global__ void edge_exp_k(const int* __restrict__ ei, const float* __restrict__ logits,
                           const unsigned int* __restrict__ maxkey,
                           float* __restrict__ expv, float* __restrict__ sums) {
    int e = blockIdx.x * blockDim.x + threadIdx.x;
    if (e >= E_EDGES) return;
    int dst = ei[E_EDGES + e];
    unsigned key = maxkey[dst];
    unsigned ib = (key & 0x80000000u) ? (key & 0x7FFFFFFFu) : ~key;
    float mx = __uint_as_float(ib);
    float ex = __expf(logits[e] - mx);
    expv[e] = ex;
    atomicAdd(&sums[dst], ex);
}

// ---------------- pass C: alpha, deg, scatter alpha*xl[src] into h ----------------
__global__ void edge_alpha_agg_k(const int* __restrict__ ei, const float* __restrict__ expv,
                                 const float* __restrict__ sums, const float* __restrict__ xl,
                                 float* __restrict__ alpha_out, float* __restrict__ deg,
                                 float* __restrict__ h) {
    int w = blockIdx.x * (blockDim.x >> 5) + (threadIdx.x >> 5);
    if (w >= E_EDGES) return;
    int lane = threadIdx.x & 31;
    int src = ei[w];
    int dst = ei[E_EDGES + w];
    float alpha = expv[w] / (sums[dst] + 1e-16f);
    if (lane == 0) {
        alpha_out[w] = alpha;
        atomicAdd(&deg[dst], alpha);
    }
    float4 a = ((const float4*)(xl + (size_t)src * C_DIM))[lane];
    float* hp = h + (size_t)dst * C_DIM + lane * 4;
    atomicAdd(hp + 0, alpha * a.x);
    atomicAdd(hp + 1, alpha * a.y);
    atomicAdd(hp + 2, alpha * a.z);
    atomicAdd(hp + 3, alpha * a.w);
}

// ---------------- relu(h + gat_bias) -> f16 ----------------
__global__ void relu_cvt_k(const float* __restrict__ h, const float* __restrict__ gb,
                           _Float16* __restrict__ hf) {
    int i = blockIdx.x * blockDim.x + threadIdx.x;
    if (i >= N_NODES * C_DIM) return;
    float v = h[i] + gb[i & (C_DIM - 1)];
    v = v > 0.f ? v : 0.f;
    hf[i] = (_Float16)v;
}

// ---------------- dis = deg > 0 ? rsqrt(deg) : 0 ----------------
__global__ void deg_inv_k(const float* __restrict__ deg, float* __restrict__ dis) {
    int n = blockIdx.x * blockDim.x + threadIdx.x;
    if (n >= N_NODES) return;
    float d = deg[n];
    dis[n] = d > 0.f ? rsqrtf(d) : 0.f;
}

// ---------------- pack Wg (f32 row-major [K=128][N=128]) into WMMA B lane layout ----------------
// Bpk index = ((nt*4 + kc)*32 + lane)*16 + idx ; lane holds column n = nt*16 + (lane&15),
// 16 contiguous K values: k = kc*32 + (lane>=16 ? 16 : 0) + idx
__global__ void pack_B_k(const float* __restrict__ Wg, _Float16* __restrict__ Bpk) {
    int i = blockIdx.x * blockDim.x + threadIdx.x;
    if (i >= C_DIM * C_DIM) return;
    int idx  = i & 15;
    int lane = (i >> 4) & 31;
    int kc   = (i >> 9) & 3;
    int nt   = i >> 11;
    int n = nt * 16 + (lane & 15);
    int k = kc * 32 + ((lane >> 4) << 4) + idx;
    Bpk[i] = (_Float16)Wg[k * C_DIM + n];
}

// ---------------- hg = h_f16 @ Wg_f16 via v_wmma_f32_16x16x32_f16 ----------------
__global__ void gemm_wmma_k(const _Float16* __restrict__ A,   // [N,128] row-major f16
                            const _Float16* __restrict__ Bpk, // packed B operand
                            float* __restrict__ D) {          // [N,128] f32
    int wid = blockIdx.x * (blockDim.x >> 5) + (threadIdx.x >> 5);
    if (wid >= (N_NODES / 16)) return;       // wave-uniform exit (EXEC all-ones for WMMA)
    int lane = threadIdx.x & 31;
    int m0 = wid * 16;
    int row = m0 + (lane & 15);
    int khalf = lane >> 4;

    // A operand: lane holds row M=lane&15; 16 contiguous K = K[khalf*16 .. +15]
    v16h a[4];
#pragma unroll
    for (int kc = 0; kc < 4; ++kc) {
        const _Float16* p = A + (size_t)row * C_DIM + kc * 32 + khalf * 16;
        union { v16h v; uint4 u[2]; } t;
        t.u[0] = *(const uint4*)(p);
        t.u[1] = *(const uint4*)(p + 8);
        a[kc] = t.v;
    }

#pragma unroll
    for (int nt = 0; nt < 8; ++nt) {
        v8f c = {0.f, 0.f, 0.f, 0.f, 0.f, 0.f, 0.f, 0.f};
#pragma unroll
        for (int kc = 0; kc < 4; ++kc) {
            const _Float16* bp = Bpk + (((size_t)nt * 4 + kc) * 32 + lane) * 16;
            union { v16h v; uint4 u[2]; } b;
            b.u[0] = *(const uint4*)(bp);
            b.u[1] = *(const uint4*)(bp + 8);
            c = __builtin_amdgcn_wmma_f32_16x16x32_f16(false, a[kc], false, b.v,
                                                       (short)0, c, false, false);
        }
        // D layout: VGPR r -> M = r + 8*(lane>=16), N = lane&15
        int mlo = (lane >> 4) << 3;
        int col = nt * 16 + (lane & 15);
#pragma unroll
        for (int r = 0; r < 8; ++r) {
            D[(size_t)(m0 + mlo + r) * C_DIM + col] = c[r];
        }
    }
}

// ---------------- GCN scatter: h2[dst] += dis[src]*alpha*dis[dst] * hg[src] ----------------
__global__ void gcn_scatter_k(const int* __restrict__ ei, const float* __restrict__ alpha,
                              const float* __restrict__ dis, const float* __restrict__ hg,
                              float* __restrict__ h2) {
    int w = blockIdx.x * (blockDim.x >> 5) + (threadIdx.x >> 5);
    if (w >= E_EDGES) return;
    int lane = threadIdx.x & 31;
    int src = ei[w];
    int dst = ei[E_EDGES + w];
    float norm = dis[src] * alpha[w] * dis[dst];
    if (norm == 0.f) return;                 // wave-uniform skip
    float4 g = ((const float4*)(hg + (size_t)src * C_DIM))[lane];
    float* p = h2 + (size_t)dst * C_DIM + lane * 4;
    atomicAdd(p + 0, norm * g.x);
    atomicAdd(p + 1, norm * g.y);
    atomicAdd(p + 2, norm * g.z);
    atomicAdd(p + 3, norm * g.w);
}

// ---------------- head: out = relu(h2 + bg) @ Wo + bo ----------------
__global__ void head_k(const float* __restrict__ h2, const float* __restrict__ bg,
                       const float* __restrict__ Wo, const float* __restrict__ bo,
                       float* __restrict__ out) {
    int n = blockIdx.x * (blockDim.x >> 5) + (threadIdx.x >> 5);
    if (n >= N_NODES) return;
    int lane = threadIdx.x & 31;
    float4 v = ((const float4*)(h2 + (size_t)n * C_DIM))[lane];
    float4 b = ((const float4*)bg)[lane];
    v.x += b.x; v.y += b.y; v.z += b.z; v.w += b.w;
    v.x = v.x > 0.f ? v.x : 0.f; v.y = v.y > 0.f ? v.y : 0.f;
    v.z = v.z > 0.f ? v.z : 0.f; v.w = v.w > 0.f ? v.w : 0.f;
    int c0 = lane * 4;
    float o0 = v.x * Wo[(c0 + 0) * 2] + v.y * Wo[(c0 + 1) * 2]
             + v.z * Wo[(c0 + 2) * 2] + v.w * Wo[(c0 + 3) * 2];
    float o1 = v.x * Wo[(c0 + 0) * 2 + 1] + v.y * Wo[(c0 + 1) * 2 + 1]
             + v.z * Wo[(c0 + 2) * 2 + 1] + v.w * Wo[(c0 + 3) * 2 + 1];
    o0 = wave_sum(o0);
    o1 = wave_sum(o1);
    if (lane == 0) {
        out[n * 2 + 0] = o0 + bo[0];
        out[n * 2 + 1] = o1 + bo[1];
    }
}

static inline size_t align256(size_t x) { return (x + 255) & ~(size_t)255; }

extern "C" void kernel_launch(void* const* d_in, const int* in_sizes, int n_in,
                              void* d_out, int out_size, void* d_ws, size_t ws_size,
                              hipStream_t stream) {
    const float* x   = (const float*)d_in[0];
    const int*   ei  = (const int*)d_in[1];
    const float* ef  = (const float*)d_in[2];
    const float* Wl  = (const float*)d_in[3];
    const float* bl  = (const float*)d_in[4];
    const float* Wr  = (const float*)d_in[5];
    const float* br  = (const float*)d_in[6];
    const float* We  = (const float*)d_in[7];
    const float* att = (const float*)d_in[8];
    const float* gb  = (const float*)d_in[9];
    const float* Wg  = (const float*)d_in[10];
    const float* bg  = (const float*)d_in[11];
    const float* Wo  = (const float*)d_in[12];
    const float* bo  = (const float*)d_in[13];

    float* out_head  = (float*)d_out;                         // [N,2]
    char*  out_ei    = (char*)d_out + (size_t)2 * N_NODES * 4; // edge_index bits
    float* out_alpha = (float*)d_out + (size_t)2 * N_NODES + (size_t)2 * E_EDGES; // [E]

    // workspace carve-up
    char* p = (char*)d_ws;
    float*     buf0   = (float*)p;      p += align256((size_t)N_NODES * C_DIM * 4); // xl -> hg
    float*     buf1   = (float*)p;      p += align256((size_t)N_NODES * C_DIM * 4); // xr -> h -> h2
    _Float16*  hf16   = (_Float16*)p;   p += align256((size_t)N_NODES * C_DIM * 2);
    _Float16*  Bpk    = (_Float16*)p;   p += align256((size_t)C_DIM * C_DIM * 2);
    float*     logits = (float*)p;      p += align256((size_t)E_EDGES * 4);
    float*     expv   = (float*)p;      p += align256((size_t)E_EDGES * 4);
    unsigned*  maxkey = (unsigned*)p;   p += align256((size_t)N_NODES * 4);
    float*     sums   = (float*)p;      p += align256((size_t)N_NODES * 4);
    float*     deg    = (float*)p;      p += align256((size_t)N_NODES * 4);
    float*     dis    = (float*)p;      p += align256((size_t)N_NODES * 4);

    const int edge_wave_blocks = E_EDGES / 8;          // 8 waves of 32 per 256-thr block
    const int node_wave_blocks = N_NODES / 8;

    // init accumulators (every call — graph replay determinism)
    hipMemsetAsync(sums, 0, (size_t)N_NODES * 4, stream);
    hipMemsetAsync(deg,  0, (size_t)N_NODES * 4, stream);
    init_maxkey_k<<<(N_NODES + 255) / 256, 256, 0, stream>>>(maxkey, N_NODES);

    // 1. node projections
    node_proj_k<<<N_NODES, C_DIM, 0, stream>>>(x, Wl, bl, Wr, br, buf0, buf1);

    // 2. edge logits + segment max  (reads buf0=xl, buf1=xr)
    edge_logits_k<<<edge_wave_blocks, 256, 0, stream>>>(ei, ef, buf0, buf1, We, att,
                                                        logits, maxkey);

    // buf1 (xr) dead -> becomes h accumulator
    hipMemsetAsync(buf1, 0, (size_t)N_NODES * C_DIM * 4, stream);

    // 3. exp + segment sum
    edge_exp_k<<<(E_EDGES + 255) / 256, 256, 0, stream>>>(ei, logits, maxkey, expv, sums);

    // 4. alpha + GAT aggregation into h (buf1); alpha written straight into d_out
    edge_alpha_agg_k<<<edge_wave_blocks, 256, 0, stream>>>(ei, expv, sums, buf0,
                                                           out_alpha, deg, buf1);

    // 5. relu(h+bias) -> f16 ; deg^{-1/2}
    relu_cvt_k<<<(N_NODES * C_DIM + 255) / 256, 256, 0, stream>>>(buf1, gb, hf16);
    deg_inv_k<<<(N_NODES + 255) / 256, 256, 0, stream>>>(deg, dis);

    // 6. pack Wg into WMMA B-operand layout; GEMM hg = h_f16 @ Wg  (buf0 reused as hg)
    pack_B_k<<<(C_DIM * C_DIM + 255) / 256, 256, 0, stream>>>(Wg, Bpk);
    gemm_wmma_k<<<(N_NODES / 16 + 7) / 8, 256, 0, stream>>>(hf16, Bpk, buf0);

    // buf1 (h) dead -> becomes h2 accumulator
    hipMemsetAsync(buf1, 0, (size_t)N_NODES * C_DIM * 4, stream);

    // 7. GCN scatter
    gcn_scatter_k<<<edge_wave_blocks, 256, 0, stream>>>(ei, out_alpha, dis, buf0, buf1);

    // 8. head
    head_k<<<node_wave_blocks, 256, 0, stream>>>(buf1, bg, Wo, bo, out_head);

    // 9. edge_index passthrough (raw int bits into float-typed d_out region)
    hipMemcpyAsync(out_ei, d_in[1], (size_t)2 * E_EDGES * sizeof(int),
                   hipMemcpyDeviceToDevice, stream);
}
